// HGNNExpertCoupler_84705345012273
// MI455X (gfx1250) — matrix operand order repaired and checked by reading, not compiled, not run on hardware
//
#include <hip/hip_runtime.h>
#include <hip/hip_bf16.h>

// ---------------------------------------------------------------------------
// Fused HGNN expert coupler for gfx1250 (MI455X).
//
// all_pairs hyperedges on E=8 collapse analytically:
//   layer:  out_i = (6*x_i + S)/14 + b,   x = h @ W^T, S = sum_e x_e
//   pooled = S2/8 + b2  (mean over experts of layer-2 output)
// Three GEMMs (~73 GFLOP) -> f16 WMMA w/ f32 accumulate, fully fused:
// HBM traffic 128MB in (NT) + 16MB out (NT) ~ 6us floor @ 23.3 TB/s;
// compute bound at f16 WMMA rate. Target <=256 VGPRs so 2 WGs (274KB LDS)
// co-reside per WGP -> 4 waves/SIMD.
// ---------------------------------------------------------------------------

typedef __attribute__((ext_vector_type(16))) _Float16 v16h;
typedef __attribute__((ext_vector_type(8)))  float    v8f;
typedef __attribute__((ext_vector_type(4)))  float    v4f;

#define HP   520              // h tile row pitch in halves (512 + 8 pad)
#define WP   72               // W panel row pitch in halves (64 + 8 pad)
#define OFFS_W 66560          // byte offset of W panel region in LDS
#define LDS_BYTES 140288      // 64*HP*2 + 512*WP*2

union Frag { uint4 q[2]; v16h h; };

// A fragment (16x32 f16), per ISA table: lane<16 -> M=lane, K={0..7,16..23};
// lane>=16 -> M=lane-16, K={8..15,24..31}.  Two ds_load_b128 per fragment.
__device__ __forceinline__ v16h lds_load_afrag(const unsigned char* base, int row,
                                               int kb, int lane) {
  const unsigned char* p = base + (((size_t)row * HP + kb + ((lane >> 4) << 3)) << 1);
  Frag f;
  f.q[0] = *(const uint4*)p;         // K 0..7   (or 8..15)
  f.q[1] = *(const uint4*)(p + 32);  // K 16..23 (or 24..31)
  return f.h;
}

// B fragment (32x16 f16): lane<16 -> N=lane, K={0..7,16..23};
// lane>=16 -> N=lane-16, K={8..15,24..31} (matches A fragment order).
__device__ __forceinline__ v16h lds_load_bfrag(const unsigned char* wbase, int fcol,
                                               int kl, int lane) {
  const unsigned char* p = wbase + (((size_t)fcol * WP + kl + ((lane >> 4) << 3)) << 1);
  Frag f;
  f.q[0] = *(const uint4*)p;         // K 0..7   (or 8..15)
  f.q[1] = *(const uint4*)(p + 32);  // K 16..23 (or 24..31)
  return f.h;
}

// Stage a 512(f) x 64(d) f32 panel of W into LDS as f16. 256 threads.
// Unroll capped at 8 to keep staged loads from ballooning live VGPRs.
__device__ __forceinline__ void stage_w_panel(const float* __restrict__ Wsrc, int p,
                                              unsigned char* wlds, int tid) {
  const float* src = Wsrc + (p << 6);
#pragma unroll 8
  for (int i = 0; i < 32; ++i) {
    int idx = tid + (i << 8);          // 0..8191 float4s
    int f   = idx >> 4;
    int c4  = idx & 15;
    v4f v = *(const v4f*)(src + (size_t)f * 512 + (c4 << 2));
    union { _Float16 h[4]; uint2 u; } cv;
    cv.h[0] = (_Float16)v.x; cv.h[1] = (_Float16)v.y;
    cv.h[2] = (_Float16)v.z; cv.h[3] = (_Float16)v.w;
    *(uint2*)(wlds + (((size_t)f * WP + (c4 << 2)) << 1)) = cv.u;
  }
}

__global__ __launch_bounds__(256, 4)
void hgnn_coupler_fused(const float* __restrict__ x,        // [8,1024,8,512]
                        const float* __restrict__ hgnn_w,   // [2,512,512]
                        const float* __restrict__ hgnn_b,   // [2,512]
                        const float* __restrict__ comb_w,   // [512,512]
                        const float* __restrict__ comb_b,   // [512]
                        const float* __restrict__ ln_g,     // [512]
                        const float* __restrict__ ln_b,     // [512]
                        float* __restrict__ out)            // [8,1024,512]
{
  __shared__ alignas(16) unsigned char lds[LDS_BYTES];
  unsigned char* hbuf = lds;            // 64 x HP halves (h tile / pooled alias)
  unsigned char* wbuf = lds + OFFS_W;   // 512 x WP halves (W panel / zbuf alias)

  const int tid  = threadIdx.x;
  const int lane = tid & 31;
  const int wv   = tid >> 5;            // 8 waves
  const int hi   = lane >> 4;           // lane half
  const int ln16 = lane & 15;

  // ---- stage input tile: 64 rows x 512 f32 -> f16 in LDS (non-temporal) -----
  {
    const float* src = x + (size_t)blockIdx.x * 64 * 512;
#pragma unroll 8
    for (int i = 0; i < 32; ++i) {
      int idx = tid + (i << 8);        // 0..8191 float4s
      int row = idx >> 7;
      int c4  = idx & 127;
      v4f v = __builtin_nontemporal_load((const v4f*)(src + (size_t)row * 512 + (c4 << 2)));
      union { _Float16 h[4]; uint2 u; } cv;
      cv.h[0] = (_Float16)v.x; cv.h[1] = (_Float16)v.y;
      cv.h[2] = (_Float16)v.z; cv.h[3] = (_Float16)v.w;
      *(uint2*)(hbuf + (((size_t)row * HP + (c4 << 2)) << 1)) = cv.u;
    }
  }

  // ---- two hypergraph-conv layers -------------------------------------------
  for (int layer = 0; layer < 2; ++layer) {
    const float* Wl = hgnn_w + (size_t)layer * 512 * 512;
    const float* bl = hgnn_b + layer * 512;

    v8f acc[4][4];
#pragma unroll
    for (int m = 0; m < 4; ++m)
#pragma unroll
      for (int n = 0; n < 4; ++n)
        acc[m][n] = (v8f){0.f, 0.f, 0.f, 0.f, 0.f, 0.f, 0.f, 0.f};

    for (int p = 0; p < 8; ++p) {
      stage_w_panel(Wl, p, wbuf, tid);
      if (p < 7)  // pull next panel toward L2/WGP$ (global_prefetch_b8)
        __builtin_prefetch(Wl + (size_t)(tid * 2) * 512 + ((p + 1) << 6), 0, 1);
      __syncthreads();

#pragma unroll
      for (int kk = 0; kk < 2; ++kk) {
        const int kb = (p << 6) + (kk << 5);  // global K for h tile
        const int kl = kk << 5;               // local K inside W panel
        v16h a[4];
#pragma unroll
        for (int m = 0; m < 4; ++m)
          a[m] = lds_load_afrag(hbuf, (m << 4) + ln16, kb, lane);
#pragma unroll
        for (int n = 0; n < 4; ++n) {
          const int fcol = ((wv << 2) + n) * 16 + ln16;
          v16h b = lds_load_bfrag(wbuf, fcol, kl, lane);
#pragma unroll
          for (int m = 0; m < 4; ++m)
            acc[m][n] = __builtin_amdgcn_wmma_f32_16x16x32_f16(
                false, a[m], false, b, (short)0, acc[m][n], false, false);
        }
      }
      __syncthreads();
    }

    // ---- hypergraph mixing: lane-local (experts = the 8 C VGPRs) ------------
#pragma unroll
    for (int n = 0; n < 4; ++n) {
      const int f  = ((wv << 2) + n) * 16 + ln16;
      const float bv = bl[f];
#pragma unroll
      for (int m = 0; m < 4; ++m) {
        v8f c = acc[m][n];
        float S = c[0] + c[1] + c[2] + c[3] + c[4] + c[5] + c[6] + c[7];
        if (layer == 0) {
#pragma unroll
          for (int j = 0; j < 8; ++j) {
            float o = (6.0f * c[j] + S) * (1.0f / 14.0f) + bv;
            int row = (m << 4) + j + (hi << 3);
            *(_Float16*)(hbuf + (((size_t)row * HP + f) << 1)) = (_Float16)o;
          }
        } else {
          // pooled = mean_e[(6 x + S)/14 + b] = S/8 + b ; rows 0..7 = tokens
          float pool = S * 0.125f + bv;
          int row = (m << 1) + hi;
          *(_Float16*)(hbuf + (((size_t)row * HP + f) << 1)) = (_Float16)pool;
        }
      }
    }
    // visibility of hbuf writes is guaranteed by the __syncthreads() that
    // follows the next phase's stage_w_panel before anyone reads hbuf.
  }

  // ---- combiner GEMM: pooled[8,512] @ comb_w^T ------------------------------
  v8f cacc[4];
#pragma unroll
  for (int n = 0; n < 4; ++n)
    cacc[n] = (v8f){0.f, 0.f, 0.f, 0.f, 0.f, 0.f, 0.f, 0.f};

  for (int p = 0; p < 8; ++p) {
    stage_w_panel(comb_w, p, wbuf, tid);
    if (p < 7)
      __builtin_prefetch(comb_w + (size_t)(tid * 2) * 512 + ((p + 1) << 6), 0, 1);
    __syncthreads();
#pragma unroll
    for (int kk = 0; kk < 2; ++kk) {
      const int kb = (p << 6) + (kk << 5);
      v16h a = lds_load_afrag(hbuf, ln16, kb, lane);   // rows 8..15 = don't-care
#pragma unroll
      for (int n = 0; n < 4; ++n) {
        const int fcol = ((wv << 2) + n) * 16 + ln16;
        v16h b = lds_load_bfrag(wbuf, fcol, kk << 5, lane);
        cacc[n] = __builtin_amdgcn_wmma_f32_16x16x32_f16(
            false, a, false, b, (short)0, cacc[n], false, false);
      }
    }
    __syncthreads();
  }

  // ---- exact GELU, stash z (f32) in LDS (aliases dead W panel) --------------
  float* zb = (float*)(lds + OFFS_W);   // [8 tokens][512]
  if (hi == 0) {                         // valid C rows: M=0..7 = tokens 0..7
#pragma unroll
    for (int n = 0; n < 4; ++n) {
      const int f  = ((wv << 2) + n) * 16 + ln16;
      const float cb = comb_b[f];
#pragma unroll
      for (int j = 0; j < 8; ++j) {
        float v = cacc[n][j] + cb;
        float z = 0.5f * v * (1.0f + erff(v * 0.70710678118654752f));
        zb[(size_t)j * 512 + f] = z;
      }
    }
  }
  __syncthreads();

  // ---- LayerNorm: wave wv handles token wv ----------------------------------
  {
    const int tok = wv;
    const float* zrow = zb + (size_t)tok * 512 + (lane << 4);
    float zv[16];
    float s = 0.f, sq = 0.f;
#pragma unroll
    for (int i = 0; i < 16; ++i) {
      float z = zrow[i];
      zv[i] = z;
      s += z;
      sq += z * z;
    }
#pragma unroll
    for (int off = 16; off >= 1; off >>= 1) {   // wave32 butterfly
      s  += __shfl_xor(s,  off, 32);
      sq += __shfl_xor(sq, off, 32);
    }
    const float mu  = s * (1.0f / 512.0f);
    const float var = sq * (1.0f / 512.0f) - mu * mu;
    const float rs  = rsqrtf(var + 1e-5f);

    const size_t obase = ((size_t)blockIdx.x * 8 + tok) * 512 + (lane << 4);
    const v4f* g4 = (const v4f*)(ln_g + (lane << 4));
    const v4f* b4 = (const v4f*)(ln_b + (lane << 4));
    v4f* o4 = (v4f*)(out + obase);
#pragma unroll
    for (int q = 0; q < 4; ++q) {
      v4f g = g4[q], bb = b4[q], r;
      r.x = (zv[q * 4 + 0] - mu) * rs * g.x + bb.x;
      r.y = (zv[q * 4 + 1] - mu) * rs * g.y + bb.y;
      r.z = (zv[q * 4 + 2] - mu) * rs * g.z + bb.z;
      r.w = (zv[q * 4 + 3] - mu) * rs * g.w + bb.w;
      __builtin_nontemporal_store(r, o4 + q);   // single-use output stream
    }
  }
}

extern "C" void kernel_launch(void* const* d_in, const int* in_sizes, int n_in,
                              void* d_out, int out_size, void* d_ws, size_t ws_size,
                              hipStream_t stream) {
  (void)in_sizes; (void)n_in; (void)out_size; (void)d_ws; (void)ws_size;
  const float* x      = (const float*)d_in[0];  // expert_outputs
  const float* hgnn_w = (const float*)d_in[1];
  const float* hgnn_b = (const float*)d_in[2];
  const float* comb_w = (const float*)d_in[3];
  const float* comb_b = (const float*)d_in[4];
  const float* ln_g   = (const float*)d_in[5];
  const float* ln_bt  = (const float*)d_in[6];
  // d_in[7]/d_in[8] (nodes_idx/edges_idx) are static all_pairs -> folded into
  // the closed-form (6x+S)/14 mixing.

  // 8192 tokens / 8 tokens-per-WG = 1024 workgroups, 256 threads (8 wave32s).
  hgnn_coupler_fused<<<dim3(1024), dim3(256), 0, stream>>>(
      x, hgnn_w, hgnn_b, comb_w, comb_b, ln_g, ln_bt, (float*)d_out);
}
